// DGCNN_32177894982305
// MI455X (gfx1250) — matrix operand order repaired and checked by reading, not compiled, not run on hardware
//
#include <hip/hip_runtime.h>
#include <hip/hip_bf16.h>
#include <math.h>

typedef __attribute__((ext_vector_type(16))) _Float16 v16h;
typedef __attribute__((ext_vector_type(8)))  float    v8f;

#define BG      128      // graphs
#define NN      1024     // nodes per graph
#define EGE     16384    // edges per graph
#define FIN     64
#define HIDC    128
#define C1C     64
#define C2C     32
#define NCLS    4

// ---------------------------------------------------------------------------
// K0: power iteration for lambda_max (raw weights) + relu-degree vector.
// One workgroup, thread i owns node i. LDS float atomics for segment sums.
// ---------------------------------------------------------------------------
__global__ __launch_bounds__(1024)
void k_lambda(const int* __restrict__ ei, const float* __restrict__ w,
              float* __restrict__ scale_out, float* __restrict__ deg_relu_out) {
    __shared__ float v[NN];
    __shared__ float agg[NN];
    __shared__ float deg[NN];
    __shared__ float red[NN];
    const int i = threadIdx.x;
    const int* src = ei;
    const int* dst = ei + EGE;

    deg[i] = 0.f; __syncthreads();
    for (int e = i; e < EGE; e += 1024) atomicAdd(&deg[src[e]], w[e]);
    __syncthreads();

    v[i] = 0.03125f;   // 1/sqrt(1024)
    __syncthreads();

    for (int it = 0; it < 50; ++it) {
        agg[i] = 0.f; __syncthreads();
        for (int e = i; e < EGE; e += 1024) atomicAdd(&agg[dst[e]], w[e] * v[src[e]]);
        __syncthreads();
        float u = deg[i] * v[i] - agg[i];
        red[i] = u * u; __syncthreads();
        for (int s = 512; s > 0; s >>= 1) { if (i < s) red[i] += red[i + s]; __syncthreads(); }
        float nrm = sqrtf(red[0]) + 1e-12f;
        v[i] = u / nrm;
        __syncthreads();
    }

    // lmax = v . (L v)
    agg[i] = 0.f; __syncthreads();
    for (int e = i; e < EGE; e += 1024) atomicAdd(&agg[dst[e]], w[e] * v[src[e]]);
    __syncthreads();
    float u = deg[i] * v[i] - agg[i];
    red[i] = v[i] * u; __syncthreads();
    for (int s = 512; s > 0; s >>= 1) { if (i < s) red[i] += red[i + s]; __syncthreads(); }
    if (i == 0) scale_out[0] = 2.0f / red[0];

    // degree with relu'd weights (used by lhat_mv)
    agg[i] = 0.f; __syncthreads();
    for (int e = i; e < EGE; e += 1024) atomicAdd(&agg[src[e]], fmaxf(w[e], 0.f));
    __syncthreads();
    deg_relu_out[i] = agg[i];
}

// ---------------------------------------------------------------------------
// K1: deterministic CSR (by dst) build of the shared edge structure.
// Thread n appends its row's edges in ascending edge order.
// ---------------------------------------------------------------------------
__global__ __launch_bounds__(1024)
void k_csr(const int* __restrict__ ei, const float* __restrict__ w,
           int* __restrict__ row_ptr, int* __restrict__ col, float* __restrict__ wv) {
    __shared__ int cnt[NN];
    __shared__ int rp[NN];
    __shared__ int dstc[2048];
    const int i = threadIdx.x;
    const int* src = ei;
    const int* dst = ei + EGE;

    cnt[i] = 0; __syncthreads();
    for (int e = i; e < EGE; e += 1024) atomicAdd(&cnt[dst[e]], 1);
    __syncthreads();
    if (i == 0) {
        int run = 0;
        for (int n = 0; n < NN; ++n) { rp[n] = run; run += cnt[n]; }
    }
    __syncthreads();

    int pos = rp[i];
    row_ptr[i] = pos;
    if (i == 0) row_ptr[NN] = EGE;

    for (int c = 0; c < EGE; c += 2048) {
        __syncthreads();
        for (int t = i; t < 2048; t += 1024) dstc[t] = dst[c + t];
        __syncthreads();
        for (int t = 0; t < 2048; ++t) {
            if (dstc[t] == i) {
                int e = c + t;
                col[pos] = src[e];
                wv[pos]  = fmaxf(w[e], 0.f);
                ++pos;
            }
        }
    }
}

// ---------------------------------------------------------------------------
// K_prep: pre-swizzle GEMM weights into per-lane f16 WMMA B-fragment order.
//   pB1[ks][nw][lane][e]  (6 x 8 x 32 x 16 halves) for the Cheb GEMM
//   pB2[ks][nw][lane][e]  (4 x 4 x 32 x 16 halves) for the conv1d GEMM
// B-frag layout: n = nw*16 + (lane&15), k = ks*32 + (lane>>4)*8 + (e&7) + (e>>3)*16.
// ---------------------------------------------------------------------------
__global__ __launch_bounds__(256)
void k_prep_w(const float* __restrict__ chebW, const float* __restrict__ convW,
              _Float16* __restrict__ pB1, _Float16* __restrict__ pB2) {
    const int idx = blockIdx.x * 256 + threadIdx.x;
    if (idx < 6 * 8 * 32 * 16) {
        const int e = idx & 15;
        const int lane = (idx >> 4) & 31;
        const int nw = (idx >> 9) & 7;
        const int ks = idx >> 12;
        const int k = ks * 32 + (lane >> 4) * 8 + (e & 7) + (e >> 3) * 16;
        const int n = nw * 16 + (lane & 15);
        pB1[idx] = (_Float16)chebW[(k >> 6) * (FIN * HIDC) + (k & 63) * HIDC + n];
    }
    if (idx < 4 * 4 * 32 * 16) {
        const int e = idx & 15;
        const int lane = (idx >> 4) & 31;
        const int nw = (idx >> 9) & 3;
        const int ks = idx >> 11;
        const int k = ks * 32 + (lane >> 4) * 8 + (e & 7) + (e >> 3) * 16;
        const int n = nw * 16 + (lane & 15);
        pB2[idx] = (_Float16)convW[n * HIDC + k];
    }
}

// ---------------------------------------------------------------------------
// K2/K3: gather-form SpMM + Chebyshev update.
//   out = alpha * ( scale*(deg*P - A@P) - P ) + beta * X0
// tx1: P=x,  alpha=1, beta= 0
// tx2: P=tx1,alpha=2, beta=-1, X0=x
// One block per (graph, 128-node chunk); 2 threads per node (32 feats each).
// ---------------------------------------------------------------------------
__global__ __launch_bounds__(256)
void k_spmm(const float* __restrict__ P, const float* __restrict__ X0,
            const int* __restrict__ row_ptr, const int* __restrict__ col,
            const float* __restrict__ wv, const float* __restrict__ deg,
            const float* __restrict__ scale_p, float* __restrict__ out,
            float alpha, float beta) {
    const int blk = blockIdx.x;          // 0..1023
    const int b = blk >> 3;
    const int nchunk = blk & 7;
    const int t = threadIdx.x;
    const int nl = t >> 1;               // 0..127
    const int hp = t & 1;                // feature half
    const int n = nchunk * 128 + nl;
    const int r = b * NN + n;

    const float sc = scale_p[0];
    const float dn = deg[n];

    float4 acc[8];
#pragma unroll
    for (int q = 0; q < 8; ++q) acc[q] = make_float4(0.f, 0.f, 0.f, 0.f);

    const int beg = row_ptr[n], end = row_ptr[n + 1];
    for (int j = beg; j < end; ++j) {
        const int s = col[j];
        const float wj = wv[j];
        const float4* xs = (const float4*)(P + (b * NN + s) * FIN) + hp * 8;
#pragma unroll
        for (int q = 0; q < 8; ++q) {
            float4 xv = xs[q];
            acc[q].x += wj * xv.x; acc[q].y += wj * xv.y;
            acc[q].z += wj * xv.z; acc[q].w += wj * xv.w;
        }
    }

    const float4* pr = (const float4*)(P  + r * FIN) + hp * 8;
    const float4* x0 = (const float4*)(X0 + r * FIN) + hp * 8;
    float4* op = (float4*)(out + r * FIN) + hp * 8;
#pragma unroll
    for (int q = 0; q < 8; ++q) {
        float4 pv = pr[q];
        float4 xv = x0[q];
        float4 o;
        o.x = alpha * (sc * (dn * pv.x - acc[q].x) - pv.x) + beta * xv.x;
        o.y = alpha * (sc * (dn * pv.y - acc[q].y) - pv.y) + beta * xv.y;
        o.z = alpha * (sc * (dn * pv.z - acc[q].z) - pv.z) + beta * xv.z;
        o.w = alpha * (sc * (dn * pv.w - acc[q].w) - pv.w) + beta * xv.w;
        op[q] = o;
    }
}

// ---------------------------------------------------------------------------
// K4: fused ChebConv GEMM + 1x1 conv1d + relu + transposed store, via WMMA.
//   H[64x128] = [tx0|tx1|tx2](64x192) @ Wcat(192x128) + cheb_b
//   Cm[64x64] = relu(H @ conv1d_W^T + conv1d_b)
//   flat[b, c*1024 + node] = Cm
// One block per (graph, 64-node tile). 256 threads = 8 waves.
// B-fragments come pre-swizzled as f16 (one 32B load per lane per kstep).
// LDS reused across phases: T tile (f16, 64x200), H tile (f16, 64x136),
// C tile (f32, 64x68).
// ---------------------------------------------------------------------------
__global__ __launch_bounds__(256)
void k_cheb_conv(const float* __restrict__ x, const float* __restrict__ tx1,
                 const float* __restrict__ tx2,
                 const _Float16* __restrict__ pB1, const _Float16* __restrict__ pB2,
                 const float* __restrict__ chebB, const float* __restrict__ convB,
                 float* __restrict__ flat) {
    __shared__ __align__(16) unsigned char smem[64 * 200 * 2];   // 25600 B
    _Float16* T = (_Float16*)smem;    // stride 200 halves (conflict-free rows)
    _Float16* H = (_Float16*)smem;    // stride 136 halves
    float*    C = (float*)smem;       // stride 68 floats
    const int LDT = 200, LDH = 136, LDC = 68;

    const int blk = blockIdx.x;       // 0..2047
    const int b = blk >> 4;
    const int nt = blk & 15;
    const int m0 = nt * 64;
    const int r0 = b * NN + m0;

    const int tid = threadIdx.x;
    const int wvid = tid >> 5;
    const int lane = tid & 31;
    const int hlf = lane >> 4;
    const int ml = lane & 15;

    // ---- phase 0: stage [64 x 192] f16 operand tile -----------------------
    for (int idx = tid; idx < 64 * 64; idx += 256) {
        const int row = idx >> 6, cc = idx & 63;
        const int base = (r0 + row) * FIN + cc;
        T[row * LDT + cc]       = (_Float16)x[base];
        T[row * LDT + 64 + cc]  = (_Float16)tx1[base];
        T[row * LDT + 128 + cc] = (_Float16)tx2[base];
    }
    __syncthreads();

    // ---- phase 1: H = T @ Wcat + cheb_b  (wave w owns N cols [16w,16w+16))
    const int n0 = wvid * 16;
    v8f acc[4];
    for (int m = 0; m < 4; ++m)
        for (int e = 0; e < 8; ++e) acc[m][e] = 0.f;

    for (int ks = 0; ks < 6; ++ks) {
        const int kb = ks * 32;
        const v16h bf = *(const v16h*)(pB1 + (((ks * 8 + wvid) * 32 + lane) << 4));
#pragma unroll
        for (int mt = 0; mt < 4; ++mt) {
            const _Float16* ar = T + (mt * 16 + ml) * LDT + kb + hlf * 8;
            v16h af;
#pragma unroll
            for (int e = 0; e < 8; ++e) { af[e] = ar[e]; af[8 + e] = ar[16 + e]; }
            acc[mt] = __builtin_amdgcn_wmma_f32_16x16x32_f16(
                false, af, false, bf, (short)0, acc[mt], false, false);
        }
    }
    const float hb = chebB[n0 + ml];
    __syncthreads();                       // all waves done reading T
    for (int mt = 0; mt < 4; ++mt)
#pragma unroll
        for (int e = 0; e < 8; ++e)
            H[(mt * 16 + hlf * 8 + e) * LDH + n0 + ml] = (_Float16)(acc[mt][e] + hb);
    __syncthreads();

    // ---- phase 2: C = relu(H @ conv1d_W^T + conv1d_b) ---------------------
    const int nt2 = wvid & 3;
    const int n0b = nt2 * 16;
    const int mtb = (wvid >> 2) * 2;
    v8f acc2[2];
    for (int m = 0; m < 2; ++m)
        for (int e = 0; e < 8; ++e) acc2[m][e] = 0.f;

    for (int ks = 0; ks < 4; ++ks) {
        const int kb = ks * 32;
        const v16h bf = *(const v16h*)(pB2 + (((ks * 4 + nt2) * 32 + lane) << 4));
#pragma unroll
        for (int u = 0; u < 2; ++u) {
            const int mt = mtb + u;
            const _Float16* ar = H + (mt * 16 + ml) * LDH + kb + hlf * 8;
            v16h af;
#pragma unroll
            for (int e = 0; e < 8; ++e) { af[e] = ar[e]; af[8 + e] = ar[16 + e]; }
            acc2[u] = __builtin_amdgcn_wmma_f32_16x16x32_f16(
                false, af, false, bf, (short)0, acc2[u], false, false);
        }
    }
    const float cb = convB[n0b + ml];
    __syncthreads();                       // all waves done reading H
    for (int u = 0; u < 2; ++u) {
        const int mt = mtb + u;
#pragma unroll
        for (int e = 0; e < 8; ++e)
            C[(mt * 16 + hlf * 8 + e) * LDC + n0b + ml] = fmaxf(acc2[u][e] + cb, 0.f);
    }
    __syncthreads();

    // ---- phase 3: coalesced transposed store: flat[b, c*1024 + node] ------
    const int cch = tid >> 2;
    const int q = tid & 3;
    float* op = flat + b * (C1C * NN) + cch * NN + m0 + q * 16;
#pragma unroll
    for (int iy = 0; iy < 16; ++iy) op[iy] = C[(q * 16 + iy) * LDC + cch];
}

// ---------------------------------------------------------------------------
// K5: fc1 split-K partials. Block kc handles K slice [kc*1024, kc*1024+1024).
// flat and fc1_W are each read exactly once across the grid.
// ---------------------------------------------------------------------------
__global__ __launch_bounds__(256)
void k_fc1(const float* __restrict__ flat, const float* __restrict__ fc1W,
           float* __restrict__ partial) {
    __shared__ float fl[8 * 1024];    // 32 KB: 8 batch rows x 1024 K
    const int kc = blockIdx.x;        // 0..63
    const int base = kc * 1024;
    const int tid = threadIdx.x;
    const int j = tid & 31;           // output channel
    const int bo = tid >> 5;          // 0..7

    const float* wr = fc1W + j * (C1C * NN) + base;
    for (int bch = 0; bch < 16; ++bch) {
        __syncthreads();
        for (int idx = tid; idx < 8 * 1024; idx += 256) {
            const int bb = idx >> 10, k = idx & 1023;
            fl[idx] = flat[(bch * 8 + bb) * (C1C * NN) + base + k];
        }
        __syncthreads();
        const float* fr = &fl[bo * 1024];
        float acc = 0.f;
        for (int k = 0; k < 1024; ++k) acc += fr[k] * wr[k];
        partial[kc * (BG * C2C) + (bch * 8 + bo) * C2C + j] = acc;
    }
}

// ---------------------------------------------------------------------------
// K6: reduce partials + fc1_b, fc2, softmax. Thread b owns one batch row.
// ---------------------------------------------------------------------------
__global__ __launch_bounds__(128)
void k_head(const float* __restrict__ partial, const float* __restrict__ fc1b,
            const float* __restrict__ fc2W, const float* __restrict__ fc2b,
            float* __restrict__ out) {
    const int b = threadIdx.x;
    float z[C2C];
#pragma unroll
    for (int j = 0; j < C2C; ++j) z[j] = fc1b[j];
    for (int kc = 0; kc < 64; ++kc) {
        const float* p = partial + kc * (BG * C2C) + b * C2C;
#pragma unroll
        for (int j = 0; j < C2C; ++j) z[j] += p[j];
    }
    float y[NCLS];
#pragma unroll
    for (int c = 0; c < NCLS; ++c) {
        float a = fc2b[c];
#pragma unroll
        for (int j = 0; j < C2C; ++j) a += z[j] * fc2W[c * C2C + j];
        y[c] = a;
    }
    float m = y[0];
#pragma unroll
    for (int c = 1; c < NCLS; ++c) m = fmaxf(m, y[c]);
    float s = 0.f;
#pragma unroll
    for (int c = 0; c < NCLS; ++c) { y[c] = expf(y[c] - m); s += y[c]; }
    const float inv = 1.f / s;
#pragma unroll
    for (int c = 0; c < NCLS; ++c) out[b * NCLS + c] = y[c] * inv;
}

// ---------------------------------------------------------------------------
extern "C" void kernel_launch(void* const* d_in, const int* in_sizes, int n_in,
                              void* d_out, int out_size, void* d_ws, size_t ws_size,
                              hipStream_t stream) {
    const float* x     = (const float*)d_in[0];
    const int*   ei    = (const int*)  d_in[1];
    const float* ew    = (const float*)d_in[2];
    const float* chebW = (const float*)d_in[3];
    const float* chebB = (const float*)d_in[4];
    const float* convW = (const float*)d_in[5];
    const float* convB = (const float*)d_in[6];
    const float* fc1W  = (const float*)d_in[7];
    const float* fc1b  = (const float*)d_in[8];
    const float* fc2W  = (const float*)d_in[9];
    const float* fc2b  = (const float*)d_in[10];
    float* out = (float*)d_out;

    // workspace layout (float-indexed)
    float*    W        = (float*)d_ws;
    float*    tx1      = W;                    // 8388608
    float*    tx2      = W + 8388608;          // 8388608
    float*    flat     = W + 16777216;         // 8388608
    float*    partial  = W + 25165824;         // 262144
    float*    deg_relu = W + 25427968;         // 1024
    float*    scalep   = W + 25428992;         // 1 (+pad)
    int*      row_ptr  = (int*)(W + 25429056); // 1025 (+pad)
    int*      col      = (int*)(W + 25430144); // 16384
    float*    wvv      = W + 25446528;         // 16384
    _Float16* pB1      = (_Float16*)(W + 25462912); // 24576 halves
    _Float16* pB2      = (_Float16*)(W + 25475200); // 8192 halves

    k_prep_w<<<96, 256, 0, stream>>>(chebW, convW, pB1, pB2);
    k_lambda<<<1, 1024, 0, stream>>>(ei, ew, scalep, deg_relu);
    k_csr<<<1, 1024, 0, stream>>>(ei, ew, row_ptr, col, wvv);
    k_spmm<<<1024, 256, 0, stream>>>(x,   x, row_ptr, col, wvv, deg_relu, scalep,
                                     tx1, 1.f,  0.f);
    k_spmm<<<1024, 256, 0, stream>>>(tx1, x, row_ptr, col, wvv, deg_relu, scalep,
                                     tx2, 2.f, -1.f);
    k_cheb_conv<<<2048, 256, 0, stream>>>(x, tx1, tx2, pB1, pB2, chebB, convB,
                                          flat);
    k_fc1<<<64, 256, 0, stream>>>(flat, fc1W, partial);
    k_head<<<1, 128, 0, stream>>>(partial, fc1b, fc2W, fc2b, out);
}
